// ScalableAttentionModule_39367670235461
// MI455X (gfx1250) — matrix operand order repaired and checked by reading, not compiled, not run on hardware
//
#include <hip/hip_runtime.h>

typedef __attribute__((ext_vector_type(16))) _Float16 v16h;
typedef __attribute__((ext_vector_type(8)))  float    v8f;
typedef __attribute__((ext_vector_type(4)))  unsigned int v4u;
typedef __attribute__((ext_vector_type(8)))  int      v8i;
typedef __attribute__((ext_vector_type(4)))  int      v4i;
typedef unsigned int   u32;
typedef unsigned short u16;

#define BB 4
#define NN 4096
#define DD 64
#define HH 128
#define MM 64

#if defined(__HIP_DEVICE_COMPILE__) && __has_builtin(__builtin_amdgcn_tensor_load_to_lds) && __has_builtin(__builtin_amdgcn_s_wait_tensorcnt)
#define HAVE_TDM 1
#else
#define HAVE_TDM 0
#endif

// workspace layout (in f16 elements)
#define OFF_W1T 0L          // [128][64]
#define OFF_W2T 8192L       // [64][128]
#define OFF_W3T 16384L      // [128][128]
#define OFF_W4T 32768L      // [64][128]
#define OFF_W5T 40960L      // [128][128]
#define OFF_W6T 57344L      // [64][128]
#define OFF_XH  65536L      // [B*N][64]
#define OFF_MSG 1114112L    // [B*N][64]   row-major messages (A-frag source)
#define OFF_AGG 2162688L    // [B*N][64]
#define OFF_FIN 3211264L    // [B*N][64]
#define OFF_MSGT 4259840L   // [B][64][N]  feature-major messages (B-frag source)

union Frag {
  v16h h;
  u32  u[8];
  _Float16 e[16];
};

__device__ __forceinline__ u16 f2h(float x) {
  union { _Float16 h; u16 u; } cv; cv.h = (_Float16)x; return cv.u;
}

// K offset of packed pair for A fragment (16-bit A, 16x32):
// VGPR v<4 -> K=2v ; v>=4 -> K=2v+8 ; lanes 16..31 add 8
__device__ __forceinline__ int kpairA(int v, int laneHi) {
  return (v < 4 ? 2 * v : 2 * v + 8) + (laneHi ? 8 : 0);
}
// K offset of packed pair for B fragment (16-bit B, 32x16):
// same K order as A: VGPR v<4 -> K=2v ; v>=4 -> K=2v+8 ; lanes 16..31 add 8
__device__ __forceinline__ int kpairB(int v, int laneHi) {
  return (v < 4 ? 2 * v : 2 * v + 8) + (laneHi ? 8 : 0);
}

// A fragment from row-major f16 [16 rows][K], base pre-offset to tile start
__device__ __forceinline__ v16h loadA(const u16* base, int K, int kk, int lane) {
  int laneHi = lane >> 4;
  int row = lane & 15;
  const u32* b32 = (const u32*)base;
  Frag f;
#pragma unroll
  for (int v = 0; v < 8; ++v) {
    int k = kk + kpairA(v, laneHi);
    f.u[v] = b32[(row * K + k) >> 1];
  }
  return f.h;
}

// B fragment from transposed weights f16 [out][K]
__device__ __forceinline__ v16h loadBw(const u16* wt, int K, int kk, int ct, int lane) {
  int laneHi = lane >> 4;
  int col = ct * 16 + (lane & 15);
  const u32* b32 = (const u32*)wt;
  Frag f;
#pragma unroll
  for (int v = 0; v < 8; ++v) {
    int k = kk + kpairB(v, laneHi);
    f.u[v] = b32[(col * K + k) >> 1];
  }
  return f.h;
}

// ---------------- K0: convert weights (transposed) + states to f16 ----------
__device__ __forceinline__ void transposeW(const float* W, int in, int out,
                                           u16* dst, long tid, long stride) {
  for (long i = tid; i < (long)in * out; i += stride) {
    int r = (int)(i / out), c = (int)(i % out);
    dst[(long)c * in + r] = f2h(W[i]);
  }
}

__global__ void k_convert(const float* __restrict__ X,
                          const float* __restrict__ W1, const float* __restrict__ W2,
                          const float* __restrict__ W3, const float* __restrict__ W4,
                          const float* __restrict__ W5, const float* __restrict__ W6,
                          u16* __restrict__ ws, long nX) {
  long tid = (long)blockIdx.x * blockDim.x + threadIdx.x;
  long stride = (long)gridDim.x * blockDim.x;
  for (long i = tid; i < nX; i += stride) ws[OFF_XH + i] = f2h(X[i]);
  transposeW(W1, DD, HH, ws + OFF_W1T, tid, stride);
  transposeW(W2, HH, MM, ws + OFF_W2T, tid, stride);
  transposeW(W3, 2 * MM, HH, ws + OFF_W3T, tid, stride);
  transposeW(W4, HH, MM, ws + OFF_W4T, tid, stride);
  transposeW(W5, DD + MM, HH, ws + OFF_W5T, tid, stride);
  transposeW(W6, HH, DD, ws + OFF_W6T, tid, stride);
}

// ---------------- K1: messages = relu(X@W1+b1)@W2+b2 ------------------------
// writes row-major MSG [B*N][64] and feature-major MSGT [B][64][N]
__global__ void k_mlp_msg(const u16* __restrict__ Xh,
                          const u16* __restrict__ W1t, const float* __restrict__ b1,
                          const u16* __restrict__ W2t, const float* __restrict__ b2,
                          u16* __restrict__ MSGh, u16* __restrict__ MSGTh) {
  __shared__ _Float16 hid[4][16 * HH];
  int lane = threadIdx.x & 31, wave = threadIdx.x >> 5;
  int laneHi = lane >> 4, nsub = lane & 15;
  int rowb = (blockIdx.x * 4 + wave) * 16;          // row within batch
  long base = (long)blockIdx.y * NN + rowb;          // global row
  const u16* X = Xh + base * DD;

  // layer 1: 64 -> 128 (relu)
  for (int ct = 0; ct < 8; ++ct) {
    v8f acc = {};
    float bias = b1[ct * 16 + nsub];
#pragma unroll
    for (int r = 0; r < 8; ++r) acc[r] = bias;
#pragma unroll
    for (int kt = 0; kt < 2; ++kt) {
      v16h a = loadA(X, DD, kt * 32, lane);
      v16h b = loadBw(W1t, DD, kt * 32, ct, lane);
      acc = __builtin_amdgcn_wmma_f32_16x16x32_f16(false, a, false, b, (short)0, acc, false, false);
    }
#pragma unroll
    for (int r = 0; r < 8; ++r) {
      float v = acc[r] > 0.f ? acc[r] : 0.f;
      hid[wave][(r + laneHi * 8) * HH + ct * 16 + nsub] = (_Float16)v;
    }
  }
  __syncthreads();
  // layer 2: 128 -> 64 (no relu)
  u16* out = MSGh + base * MM;
  u16* outT = MSGTh + (long)blockIdx.y * NN * MM;    // [64][N]
  for (int ct = 0; ct < 4; ++ct) {
    v8f acc = {};
    float bias = b2[ct * 16 + nsub];
#pragma unroll
    for (int r = 0; r < 8; ++r) acc[r] = bias;
#pragma unroll
    for (int kt = 0; kt < 4; ++kt) {
      v16h a = loadA((const u16*)&hid[wave][0], HH, kt * 32, lane);
      v16h b = loadBw(W2t, HH, kt * 32, ct, lane);
      acc = __builtin_amdgcn_wmma_f32_16x16x32_f16(false, a, false, b, (short)0, acc, false, false);
    }
#pragma unroll
    for (int r = 0; r < 8; ++r) {
      u16 hv = f2h(acc[r]);
      int M = r + laneHi * 8;
      out[M * MM + ct * 16 + nsub] = hv;
      outT[(long)(ct * 16 + nsub) * NN + rowb + M] = hv;
    }
  }
}

// ---------------- K2: aggregated = (adj/deg) @ messages ----------------------
// positions staged to LDS by the Tensor Data Mover; B-frags stream from MSGT.
__global__ void k_aggregate(const float* __restrict__ pos, const u16* __restrict__ MSGTh,
                            const int* __restrict__ nrad, u16* __restrict__ AGGh) {
  __shared__ float posS[NN * 2];          // 32 KB
  __shared__ float degS[4][16];
  int lane = threadIdx.x & 31, wave = threadIdx.x >> 5;
  int laneHi = lane >> 4, nsub = lane & 15;
  int b = blockIdx.y;
  const float* P = pos + (long)b * NN * 2;

#if HAVE_TDM
  if (wave == 0) {
    // D# group0: count=1, lds_addr, 57-bit global addr, type=2
    unsigned long long ga = (unsigned long long)(uintptr_t)P;
    v4u g0 = { 1u, (u32)(uintptr_t)posS,
               (u32)ga, (u32)((ga >> 32) & 0x01FFFFFFu) | 0x80000000u };
    // D# group1: data_size=4B(code 2), tensor_dim0=8192, tensor_dim1=1,
    //            tile_dim0=8192, tile_dim1=1, tensor_dim0_stride=8192
    v8i g1 = { (int)0x00020000, (int)0x20000000, (int)0x00010000, (int)0x20000000,
               1, 0x2000, 0, 0 };
    v4i gz4 = { 0, 0, 0, 0 };
    v8i gz8 = { 0, 0, 0, 0, 0, 0, 0, 0 };
    __builtin_amdgcn_tensor_load_to_lds(g0, g1, gz4, gz4, gz8, 0);
    __builtin_amdgcn_s_wait_tensorcnt(0);
  }
#else
  for (int i = threadIdx.x; i < NN * 2; i += blockDim.x) posS[i] = P[i];
#endif
  __syncthreads();

  float rr = (float)nrad[0];
  float r2 = rr * rr;

  int rowbase = blockIdx.x * 64 + wave * 16;
  int myrow = rowbase + nsub;
  float px = posS[myrow * 2], py = posS[myrow * 2 + 1];

  v8f acc0 = {}, acc1 = {}, acc2 = {}, acc3 = {};
  float degacc = 0.f;
  const u16* MT = MSGTh + (long)b * NN * MM;   // [64][N]
  const u32* mt32 = (const u32*)MT;

  for (int jt = 0; jt < NN / 32; ++jt) {
    int jbase = jt * 32;

    // adjacency A fragment computed in-register from LDS positions
    Frag fa;
#pragma unroll
    for (int v = 0; v < 8; ++v) {
      int k = kpairA(v, laneHi);
      int c0 = jbase + k, c1 = c0 + 1;
      float dx0 = posS[c0 * 2] - px, dy0 = posS[c0 * 2 + 1] - py;
      float dx1 = posS[c1 * 2] - px, dy1 = posS[c1 * 2 + 1] - py;
      float d0 = dx0 * dx0 + dy0 * dy0;
      float d1 = dx1 * dx1 + dy1 * dy1;
      float a0 = (d0 < r2 && c0 != myrow) ? 1.f : 0.f;
      float a1 = (d1 < r2 && c1 != myrow) ? 1.f : 0.f;
      degacc += a0 + a1;
      fa.e[2 * v] = (_Float16)a0;
      fa.e[2 * v + 1] = (_Float16)a1;
    }
    if (jt + 1 < NN / 32)
      __builtin_prefetch(MT + (long)lane * NN + jbase + 32, 0, 0);

    v8f* accs[4] = {&acc0, &acc1, &acc2, &acc3};
#pragma unroll
    for (int ct = 0; ct < 4; ++ct) {
      Frag fb;   // contiguous K-pairs in MSGT -> coalesces to global_load_b128
#pragma unroll
      for (int v = 0; v < 8; ++v) {
        int k = kpairB(v, laneHi);
        fb.u[v] = mt32[((ct * 16 + nsub) * NN + jbase + k) >> 1];
      }
      *accs[ct] = __builtin_amdgcn_wmma_f32_16x16x32_f16(false, fa.h, false, fb.h, (short)0, *accs[ct], false, false);
    }
  }

  degacc += __shfl_xor(degacc, 16, 32);
  if (laneHi == 0) degS[wave][nsub] = degacc;
  __syncthreads();

  u16* out = AGGh + ((long)b * NN + rowbase) * MM;
  v8f* accs[4] = {&acc0, &acc1, &acc2, &acc3};
#pragma unroll
  for (int ct = 0; ct < 4; ++ct)
#pragma unroll
    for (int r = 0; r < 8; ++r) {
      int M = r + laneHi * 8;
      float dg = degS[wave][M];
      if (dg < 1.f) dg = 1.f;
      out[M * MM + ct * 16 + nsub] = f2h((*accs[ct])[r] / dg);
    }
}

// ------- K3: final = relu(msg@W3a + agg@W3b + b3)@W4 + b4 (split-K concat) ---
__global__ void k_mlp_final(const u16* __restrict__ MSGh, const u16* __restrict__ AGGh,
                            const u16* __restrict__ W3t, const float* __restrict__ b3,
                            const u16* __restrict__ W4t, const float* __restrict__ b4,
                            u16* __restrict__ FINh) {
  __shared__ _Float16 hid[4][16 * HH];
  int lane = threadIdx.x & 31, wave = threadIdx.x >> 5;
  int laneHi = lane >> 4, nsub = lane & 15;
  long base = (long)blockIdx.y * NN + (blockIdx.x * 4 + wave) * 16;
  const u16* A0 = MSGh + base * MM;
  const u16* A1 = AGGh + base * MM;

  for (int ct = 0; ct < 8; ++ct) {
    v8f acc = {};
    float bias = b3[ct * 16 + nsub];
#pragma unroll
    for (int r = 0; r < 8; ++r) acc[r] = bias;
#pragma unroll
    for (int kt = 0; kt < 4; ++kt) {
      v16h a = (kt < 2) ? loadA(A0, MM, kt * 32, lane)
                        : loadA(A1, MM, (kt - 2) * 32, lane);
      v16h b = loadBw(W3t, 2 * MM, kt * 32, ct, lane);
      acc = __builtin_amdgcn_wmma_f32_16x16x32_f16(false, a, false, b, (short)0, acc, false, false);
    }
#pragma unroll
    for (int r = 0; r < 8; ++r) {
      float v = acc[r] > 0.f ? acc[r] : 0.f;
      hid[wave][(r + laneHi * 8) * HH + ct * 16 + nsub] = (_Float16)v;
    }
  }
  __syncthreads();
  u16* out = FINh + base * MM;
  for (int ct = 0; ct < 4; ++ct) {
    v8f acc = {};
    float bias = b4[ct * 16 + nsub];
#pragma unroll
    for (int r = 0; r < 8; ++r) acc[r] = bias;
#pragma unroll
    for (int kt = 0; kt < 4; ++kt) {
      v16h a = loadA((const u16*)&hid[wave][0], HH, kt * 32, lane);
      v16h b = loadBw(W4t, HH, kt * 32, ct, lane);
      acc = __builtin_amdgcn_wmma_f32_16x16x32_f16(false, a, false, b, (short)0, acc, false, false);
    }
#pragma unroll
    for (int r = 0; r < 8; ++r)
      out[(r + laneHi * 8) * MM + ct * 16 + nsub] = f2h(acc[r]);
  }
}

// ------- K4: out = X + (relu(X@W5a + fin@W5b + b5)@W6 + b6)  (f32 output) ----
__global__ void k_mlp_out(const u16* __restrict__ Xh, const u16* __restrict__ FINh,
                          const float* __restrict__ Xf,
                          const u16* __restrict__ W5t, const float* __restrict__ b5,
                          const u16* __restrict__ W6t, const float* __restrict__ b6,
                          float* __restrict__ dout) {
  __shared__ _Float16 hid[4][16 * HH];
  int lane = threadIdx.x & 31, wave = threadIdx.x >> 5;
  int laneHi = lane >> 4, nsub = lane & 15;
  long base = (long)blockIdx.y * NN + (blockIdx.x * 4 + wave) * 16;
  const u16* A0 = Xh + base * DD;
  const u16* A1 = FINh + base * MM;

  for (int ct = 0; ct < 8; ++ct) {
    v8f acc = {};
    float bias = b5[ct * 16 + nsub];
#pragma unroll
    for (int r = 0; r < 8; ++r) acc[r] = bias;
#pragma unroll
    for (int kt = 0; kt < 4; ++kt) {
      v16h a = (kt < 2) ? loadA(A0, DD, kt * 32, lane)
                        : loadA(A1, MM, (kt - 2) * 32, lane);
      v16h b = loadBw(W5t, DD + MM, kt * 32, ct, lane);
      acc = __builtin_amdgcn_wmma_f32_16x16x32_f16(false, a, false, b, (short)0, acc, false, false);
    }
#pragma unroll
    for (int r = 0; r < 8; ++r) {
      float v = acc[r] > 0.f ? acc[r] : 0.f;
      hid[wave][(r + laneHi * 8) * HH + ct * 16 + nsub] = (_Float16)v;
    }
  }
  __syncthreads();
  for (int ct = 0; ct < 4; ++ct) {
    v8f acc = {};
    float bias = b6[ct * 16 + nsub];
#pragma unroll
    for (int r = 0; r < 8; ++r) acc[r] = bias;
#pragma unroll
    for (int kt = 0; kt < 4; ++kt) {
      v16h a = loadA((const u16*)&hid[wave][0], HH, kt * 32, lane);
      v16h b = loadBw(W6t, HH, kt * 32, ct, lane);
      acc = __builtin_amdgcn_wmma_f32_16x16x32_f16(false, a, false, b, (short)0, acc, false, false);
    }
#pragma unroll
    for (int r = 0; r < 8; ++r) {
      long idx = (base + r + laneHi * 8) * DD + ct * 16 + nsub;
      dout[idx] = Xf[idx] + acc[r];
    }
  }
}

extern "C" void kernel_launch(void* const* d_in, const int* in_sizes, int n_in,
                              void* d_out, int out_size, void* d_ws, size_t ws_size,
                              hipStream_t stream) {
  (void)in_sizes; (void)n_in; (void)out_size; (void)ws_size;
  const float* X   = (const float*)d_in[0];
  const float* POS = (const float*)d_in[1];
  const int*   NR  = (const int*)d_in[2];
  const float* W1 = (const float*)d_in[3];  const float* B1 = (const float*)d_in[4];
  const float* W2 = (const float*)d_in[5];  const float* B2 = (const float*)d_in[6];
  const float* W3 = (const float*)d_in[7];  const float* B3 = (const float*)d_in[8];
  const float* W4 = (const float*)d_in[9];  const float* B4 = (const float*)d_in[10];
  const float* W5 = (const float*)d_in[11]; const float* B5 = (const float*)d_in[12];
  const float* W6 = (const float*)d_in[13]; const float* B6 = (const float*)d_in[14];
  u16* ws = (u16*)d_ws;
  long nX = (long)BB * NN * DD;

  k_convert<<<1024, 256, 0, stream>>>(X, W1, W2, W3, W4, W5, W6, ws, nX);

  dim3 grid(NN / 64, BB), blk(128);
  k_mlp_msg<<<grid, blk, 0, stream>>>(ws + OFF_XH, ws + OFF_W1T, B1,
                                      ws + OFF_W2T, B2, ws + OFF_MSG, ws + OFF_MSGT);
  k_aggregate<<<grid, blk, 0, stream>>>(POS, ws + OFF_MSGT, NR, ws + OFF_AGG);
  k_mlp_final<<<grid, blk, 0, stream>>>(ws + OFF_MSG, ws + OFF_AGG,
                                        ws + OFF_W3T, B3, ws + OFF_W4T, B4,
                                        ws + OFF_FIN);
  k_mlp_out<<<grid, blk, 0, stream>>>(ws + OFF_XH, ws + OFF_FIN, X,
                                      ws + OFF_W5T, B5, ws + OFF_W6T, B6,
                                      (float*)d_out);
}
